// LSA_86131274154452
// MI455X (gfx1250) — matrix operand order repaired and checked by reading, not compiled, not run on hardware
//
#include <hip/hip_runtime.h>

// ---------------------------------------------------------------------------
// Self-attention (b=2, n=4096, dim=512, heads=8, dim_head=64) for gfx1250.
// 3 kernels, all built on v_wmma_f32_16x16x32_f16 (wave32 WMMA):
//   1) qkv_kernel : x @ w_qkv -> Q,K (f16 head-major), V (f16 d-major), Q pre-scaled
//   2) attn_kernel: flash-attention, double-buffered K/V tiles staged with
//                   GLOBAL_LOAD_ASYNC_TO_LDS (ASYNCcnt) when available
//   3) out_kernel : O @ w_out + b_out -> f32 output
// ---------------------------------------------------------------------------

typedef __attribute__((ext_vector_type(16))) _Float16 v16h;
typedef __attribute__((ext_vector_type(8)))  _Float16 v8h;
typedef __attribute__((ext_vector_type(8)))  float    v8f;
typedef __attribute__((ext_vector_type(4)))  float    v4f;
typedef __attribute__((ext_vector_type(4)))  int      v4i;

#if __has_builtin(__builtin_amdgcn_global_load_async_to_lds_b128) && \
    __has_builtin(__builtin_amdgcn_s_wait_asynccnt)
#define LSA_ASYNC_LDS 1
#else
#define LSA_ASYNC_LDS 0
#endif

#if LSA_ASYNC_LDS
// b128 async copy takes int4 pointers: global (AS1) src, LDS (AS3) dst
typedef __attribute__((address_space(1))) v4i* gvec_t;
typedef __attribute__((address_space(3))) v4i* lvec_t;
#endif

// copy 16 f16 (32B) global -> LDS
__device__ __forceinline__ void cp16_to_lds(_Float16* lds, const _Float16* g) {
#if LSA_ASYNC_LDS
  __builtin_amdgcn_global_load_async_to_lds_b128((gvec_t)g, (lvec_t)lds, 0, 0);
  __builtin_amdgcn_global_load_async_to_lds_b128((gvec_t)(g + 8),
                                                 (lvec_t)(lds + 8), 0, 0);
#else
  *(v8h*)lds = *(const v8h*)g;
  *(v8h*)(lds + 8) = *(const v8h*)(g + 8);
#endif
}

// copy 8 f16 (16B) global -> LDS
__device__ __forceinline__ void cp8_to_lds(_Float16* lds, const _Float16* g) {
#if LSA_ASYNC_LDS
  __builtin_amdgcn_global_load_async_to_lds_b128((gvec_t)g, (lvec_t)lds, 0, 0);
#else
  *(v8h*)lds = *(const v8h*)g;
#endif
}

__device__ __forceinline__ void async_wait0() {
#if LSA_ASYNC_LDS
  __builtin_amdgcn_s_wait_asynccnt(0);
#endif
}

__device__ __forceinline__ v16h mk16(v8h lo, v8h hi) {
  v16h r;
#pragma unroll
  for (int i = 0; i < 8; ++i) { r[i] = lo[i]; r[i + 8] = hi[i]; }
  return r;
}

__device__ __forceinline__ v8f wmma16(v16h a, v16h b, v8f c) {
  // D = A(16x32 f16) * B(32x16 f16) + C(16x16 f32)
  return __builtin_amdgcn_wmma_f32_16x16x32_f16(false, a, false, b,
                                                (short)0, c, false, false);
}

__device__ __forceinline__ float rowmax16(float v) {
  v = fmaxf(v, __shfl_xor(v, 1));
  v = fmaxf(v, __shfl_xor(v, 2));
  v = fmaxf(v, __shfl_xor(v, 4));
  v = fmaxf(v, __shfl_xor(v, 8));
  return v;
}

__device__ __forceinline__ float rowsum16(float v) {
  v += __shfl_xor(v, 1);
  v += __shfl_xor(v, 2);
  v += __shfl_xor(v, 4);
  v += __shfl_xor(v, 8);
  return v;
}

// ---------------------------------------------------------------------------
// Kernel 1: qkv = x @ w_qkv.
//   Q -> Qs[bh][n][d] * exp(temperature)   (f16)
//   K -> Ks[bh][n][d]                      (f16)
//   V -> Vs[bh][d][n]   (d-major so attention V tiles are contiguous copies)
// Grid: (1536/64, 8192/64), block 256 (8 waves).
// ---------------------------------------------------------------------------
__global__ __launch_bounds__(256) void qkv_kernel(
    const float* __restrict__ x, const float* __restrict__ w,
    const float* __restrict__ temp,
    _Float16* __restrict__ Qs, _Float16* __restrict__ Ks,
    _Float16* __restrict__ Vs) {
  __shared__ __align__(16) _Float16 Ash[64][40];  // [row][k] k<32
  __shared__ __align__(16) _Float16 Bsh[64][40];  // [col][k] k<32

  const int tid  = threadIdx.x;
  const int wave = tid >> 5, lane = tid & 31;
  const int hfsel = lane >> 4, l16 = lane & 15;
  const int n0 = blockIdx.x * 64;
  const int m0 = blockIdx.y * 64;
  const float scale = __expf(temp[0]);

  const int mi = wave & 3;            // 16-row subtile
  const int njb = (wave >> 2) * 2;    // 2 x 16-col subtiles

  v8f acc0 = {}, acc1 = {};

  const int ar = tid >> 2, ak = (tid & 3) * 8;   // A: 64 rows x 32 k
  const int bk = tid >> 3, bc = (tid & 7) * 8;   // B: 32 k x 64 cols

  for (int k0 = 0; k0 < 512; k0 += 32) {
    __syncthreads();
    {  // A tile: x[m0+ar][k0+ak..+7] -> f16
      const float* src = x + (size_t)(m0 + ar) * 512 + k0 + ak;
      v4f f0 = *(const v4f*)src;
      v4f f1 = *(const v4f*)(src + 4);
      v8h hh;
#pragma unroll
      for (int i = 0; i < 4; ++i) { hh[i] = (_Float16)f0[i]; hh[i + 4] = (_Float16)f1[i]; }
      *(v8h*)&Ash[ar][ak] = hh;
    }
    {  // B tile (transposed to [col][k]): w[k0+bk][n0+bc..+7]
      const float* src = w + (size_t)(k0 + bk) * 1536 + n0 + bc;
      v4f f0 = *(const v4f*)src;
      v4f f1 = *(const v4f*)(src + 4);
#pragma unroll
      for (int i = 0; i < 4; ++i) {
        Bsh[bc + i][bk]     = (_Float16)f0[i];
        Bsh[bc + 4 + i][bk] = (_Float16)f1[i];
      }
    }
    __syncthreads();

    v16h a;
    {
      const int r = mi * 16 + l16;
      a = mk16(*(const v8h*)&Ash[r][hfsel * 8],
               *(const v8h*)&Ash[r][16 + hfsel * 8]);
    }
#pragma unroll
    for (int j = 0; j < 2; ++j) {
      const int c = (njb + j) * 16 + l16;
      v16h bfr = mk16(*(const v8h*)&Bsh[c][hfsel * 8],
                      *(const v8h*)&Bsh[c][16 + hfsel * 8]);
      v8f& acc = j ? acc1 : acc0;
      acc = wmma16(a, bfr, acc);
    }
  }

#pragma unroll
  for (int j = 0; j < 2; ++j) {
    const v8f acc = j ? acc1 : acc0;
    const int gc = n0 + (njb + j) * 16 + l16;
    const int part = gc >> 9;            // 0=q 1=k 2=v
    const int within = gc & 511;
    const int hh = within >> 6, dd = within & 63;
    _Float16* dst = (part == 0) ? Qs : (part == 1) ? Ks : Vs;
    const float mult = (part == 0) ? scale : 1.0f;
#pragma unroll
    for (int i = 0; i < 8; ++i) {
      const int gr = m0 + mi * 16 + i + 8 * hfsel;       // 0..8191
      const int bb = gr >> 12, nn = gr & 4095;
      const int bhh = bb * 8 + hh;
      const size_t idx = (part == 2)
          ? ((size_t)bhh * 64 + dd) * 4096 + nn     // V: [bh][d][n]
          : ((size_t)bhh * 4096 + nn) * 64 + dd;    // Q,K: [bh][n][d]
      dst[idx] = (_Float16)(acc[i] * mult);
    }
  }
}

// ---------------------------------------------------------------------------
// Kernel 2: flash attention. One block = 128 query rows of one (b,h);
// 8 waves, each owns a 16-row strip. Key loop in tiles of 64, with
// double-buffered K/V tiles staged via async global->LDS copies.
// Grid: (4096/128, 16), block 256.
// ---------------------------------------------------------------------------
__global__ __launch_bounds__(256) void attn_kernel(
    const _Float16* __restrict__ Qg, const _Float16* __restrict__ Kg,
    const _Float16* __restrict__ Vg, _Float16* __restrict__ Og) {
  __shared__ __align__(16) _Float16 Qsh[128][72];      // [qrow][d]
  __shared__ __align__(16) _Float16 Ksh[2][64][72];    // [buf][key][d]
  __shared__ __align__(16) _Float16 Vsh[2][64][72];    // [buf][d][key]
  __shared__ __align__(16) _Float16 Psh[8][16][72];    // per-wave P scratch

  const int tid  = threadIdx.x;
  const int wave = tid >> 5, lane = tid & 31;
  const int hfsel = lane >> 4, l16 = lane & 15;
  const int bh = blockIdx.y;
  const int q0 = blockIdx.x * 128;
  const size_t base = (size_t)bh * 4096 * 64;   // same extent for Q/K/V

  const int lr = tid >> 2, ld = (tid & 3) * 16;  // tile loader mapping

  {  // stage the 128x64 Q tile (async)
    const int r = tid >> 1, ds = (tid & 1) * 32;
    const _Float16* src = Qg + base + (size_t)(q0 + r) * 64 + ds;
    cp16_to_lds(&Qsh[r][ds], src);
    cp16_to_lds(&Qsh[r][ds + 16], src + 16);
  }
  // stage tile 0 (async): K rows are keys, V rows are d (d-major layout)
  cp16_to_lds(&Ksh[0][lr][ld], Kg + base + (size_t)lr * 64 + ld);
  cp16_to_lds(&Vsh[0][lr][ld], Vg + base + (size_t)lr * 4096 + ld);

  float run_m[8], run_l[8];
  v8f o0 = {}, o1 = {}, o2 = {}, o3 = {};
#pragma unroll
  for (int i = 0; i < 8; ++i) { run_m[i] = -3.0e38f; run_l[i] = 0.0f; }

  for (int kb = 0; kb < 4096; kb += 64) {
    const int cur = (kb >> 6) & 1;
    async_wait0();     // this wave's tile-t (and Q) transfers complete
    __syncthreads();   // => everyone's transfers visible; prev compute done

    if (kb + 64 < 4096) {  // issue tile t+1 into the other buffer
      const int nxt = cur ^ 1;
      cp16_to_lds(&Ksh[nxt][lr][ld],
                  Kg + base + (size_t)(kb + 64 + lr) * 64 + ld);
      cp16_to_lds(&Vsh[nxt][lr][ld],
                  Vg + base + (size_t)lr * 4096 + kb + 64 + ld);
    }

    // ---- S = Q K^T (scale already folded into Q) ----
    v8f s0 = {}, s1 = {}, s2 = {}, s3 = {};
#pragma unroll
    for (int dstep = 0; dstep < 64; dstep += 32) {
      const int r = wave * 16 + l16;
      v16h a = mk16(*(const v8h*)&Qsh[r][dstep + hfsel * 8],
                    *(const v8h*)&Qsh[r][dstep + 16 + hfsel * 8]);
#pragma unroll
      for (int s = 0; s < 4; ++s) {
        const int c = s * 16 + l16;
        v16h bfr = mk16(*(const v8h*)&Ksh[cur][c][dstep + hfsel * 8],
                        *(const v8h*)&Ksh[cur][c][dstep + 16 + hfsel * 8]);
        v8f& sa = (s == 0) ? s0 : (s == 1) ? s1 : (s == 2) ? s2 : s3;
        sa = wmma16(a, bfr, sa);
      }
    }

    // ---- mask diagonal (q == k) ----
#pragma unroll
    for (int s = 0; s < 4; ++s) {
      v8f& sa = (s == 0) ? s0 : (s == 1) ? s1 : (s == 2) ? s2 : s3;
      const int kcol = kb + s * 16 + l16;
#pragma unroll
      for (int i = 0; i < 8; ++i) {
        const int qrow = q0 + wave * 16 + i + 8 * hfsel;
        if (qrow == kcol) sa[i] = -3.0e38f;
      }
    }

    // ---- online softmax update ----
#pragma unroll
    for (int i = 0; i < 8; ++i) {
      float m = fmaxf(fmaxf(s0[i], s1[i]), fmaxf(s2[i], s3[i]));
      m = rowmax16(m);
      m = fmaxf(m, run_m[i]);
      const float alpha = __expf(run_m[i] - m);
      const float e0 = __expf(s0[i] - m);
      const float e1 = __expf(s1[i] - m);
      const float e2 = __expf(s2[i] - m);
      const float e3 = __expf(s3[i] - m);
      s0[i] = e0; s1[i] = e1; s2[i] = e2; s3[i] = e3;
      const float rs = rowsum16(e0 + e1 + e2 + e3);
      run_l[i] = run_l[i] * alpha + rs;
      run_m[i] = m;
      o0[i] *= alpha; o1[i] *= alpha; o2[i] *= alpha; o3[i] *= alpha;
    }

    // ---- spill P (C layout) to per-wave LDS, reload as A fragments ----
#pragma unroll
    for (int s = 0; s < 4; ++s) {
      const v8f sa = (s == 0) ? s0 : (s == 1) ? s1 : (s == 2) ? s2 : s3;
#pragma unroll
      for (int i = 0; i < 8; ++i)
        Psh[wave][i + 8 * hfsel][s * 16 + l16] = (_Float16)sa[i];
    }

    // ---- O += P @ V ----
#pragma unroll
    for (int kstep = 0; kstep < 64; kstep += 32) {
      v16h a = mk16(*(const v8h*)&Psh[wave][l16][kstep + hfsel * 8],
                    *(const v8h*)&Psh[wave][l16][kstep + 16 + hfsel * 8]);
#pragma unroll
      for (int nj = 0; nj < 4; ++nj) {
        const int c = nj * 16 + l16;
        v16h bfr = mk16(*(const v8h*)&Vsh[cur][c][kstep + hfsel * 8],
                        *(const v8h*)&Vsh[cur][c][kstep + 16 + hfsel * 8]);
        v8f& oa = (nj == 0) ? o0 : (nj == 1) ? o1 : (nj == 2) ? o2 : o3;
        oa = wmma16(a, bfr, oa);
      }
    }
  }

  // ---- finalize: divide by row sums, store O as f16 [b*n][h*64+d] ----
#pragma unroll
  for (int i = 0; i < 8; ++i) {
    const float inv = 1.0f / run_l[i];
    o0[i] *= inv; o1[i] *= inv; o2[i] *= inv; o3[i] *= inv;
  }
  const int bb = bh >> 3, hh = bh & 7;
#pragma unroll
  for (int nj = 0; nj < 4; ++nj) {
    const v8f oa = (nj == 0) ? o0 : (nj == 1) ? o1 : (nj == 2) ? o2 : o3;
    const int col = hh * 64 + nj * 16 + l16;
#pragma unroll
    for (int i = 0; i < 8; ++i) {
      const int gr = bb * 4096 + q0 + wave * 16 + i + 8 * hfsel;
      Og[(size_t)gr * 512 + col] = (_Float16)oa[i];
    }
  }
}

// ---------------------------------------------------------------------------
// Kernel 3: out = O @ w_out + b_out (f32 result).
// A tile staged with async global->LDS (straight f16 copy); B needs f32->f16
// conversion so it stays on the VGPR path.
// Grid: (512/64, 8192/64), block 256.
// ---------------------------------------------------------------------------
__global__ __launch_bounds__(256) void out_kernel(
    const _Float16* __restrict__ A, const float* __restrict__ w,
    const float* __restrict__ bias, float* __restrict__ out) {
  __shared__ __align__(16) _Float16 Ash[64][40];
  __shared__ __align__(16) _Float16 Bsh[64][40];

  const int tid  = threadIdx.x;
  const int wave = tid >> 5, lane = tid & 31;
  const int hfsel = lane >> 4, l16 = lane & 15;
  const int n0 = blockIdx.x * 64;
  const int m0 = blockIdx.y * 64;

  const int mi = wave & 3;
  const int njb = (wave >> 2) * 2;

  v8f acc0 = {}, acc1 = {};

  const int ar = tid >> 2, ak = (tid & 3) * 8;
  const int bk = tid >> 3, bc = (tid & 7) * 8;

  for (int k0 = 0; k0 < 512; k0 += 32) {
    __syncthreads();
    cp8_to_lds(&Ash[ar][ak], A + (size_t)(m0 + ar) * 512 + k0 + ak);
    {
      const float* src = w + (size_t)(k0 + bk) * 512 + n0 + bc;
      v4f f0 = *(const v4f*)src;
      v4f f1 = *(const v4f*)(src + 4);
#pragma unroll
      for (int i = 0; i < 4; ++i) {
        Bsh[bc + i][bk]     = (_Float16)f0[i];
        Bsh[bc + 4 + i][bk] = (_Float16)f1[i];
      }
    }
    async_wait0();
    __syncthreads();

    v16h a;
    {
      const int r = mi * 16 + l16;
      a = mk16(*(const v8h*)&Ash[r][hfsel * 8],
               *(const v8h*)&Ash[r][16 + hfsel * 8]);
    }
#pragma unroll
    for (int j = 0; j < 2; ++j) {
      const int c = (njb + j) * 16 + l16;
      v16h bfr = mk16(*(const v8h*)&Bsh[c][hfsel * 8],
                      *(const v8h*)&Bsh[c][16 + hfsel * 8]);
      v8f& acc = j ? acc1 : acc0;
      acc = wmma16(a, bfr, acc);
    }
  }

#pragma unroll
  for (int j = 0; j < 2; ++j) {
    const v8f acc = j ? acc1 : acc0;
    const int gc = n0 + (njb + j) * 16 + l16;
    const float bv = bias[gc];
#pragma unroll
    for (int i = 0; i < 8; ++i) {
      const int gr = m0 + mi * 16 + i + 8 * hfsel;
      out[(size_t)gr * 512 + gc] = acc[i] + bv;
    }
  }
}

// ---------------------------------------------------------------------------
extern "C" void kernel_launch(void* const* d_in, const int* in_sizes, int n_in,
                              void* d_out, int out_size, void* d_ws,
                              size_t ws_size, hipStream_t stream) {
  const float* x     = (const float*)d_in[0];
  const float* w_qkv = (const float*)d_in[1];
  const float* w_out = (const float*)d_in[2];
  const float* b_out = (const float*)d_in[3];
  const float* temp  = (const float*)d_in[4];
  float* out = (float*)d_out;

  char* ws = (char*)d_ws;
  _Float16* Qs = (_Float16*)(ws);                       // 8 MiB
  _Float16* Ks = (_Float16*)(ws + ((size_t)8 << 20));   // 8 MiB
  _Float16* Vs = (_Float16*)(ws + ((size_t)16 << 20));  // 8 MiB (d-major)
  _Float16* Og = (_Float16*)(ws + ((size_t)24 << 20));  // 8 MiB

  qkv_kernel<<<dim3(1536 / 64, 8192 / 64), 256, 0, stream>>>(x, w_qkv, temp,
                                                             Qs, Ks, Vs);
  attn_kernel<<<dim3(4096 / 128, 16), 256, 0, stream>>>(Qs, Ks, Vs, Og);
  out_kernel<<<dim3(512 / 64, 8192 / 64), 256, 0, stream>>>(Og, w_out, b_out,
                                                            out);
}
